// MyBilstm_35622458753561
// MI455X (gfx1250) — matrix operand order repaired and checked by reading, not compiled, not run on hardware
//
#include <hip/hip_runtime.h>
#include <hip/hip_bf16.h>

typedef __attribute__((ext_vector_type(16))) _Float16 v16h;
typedef __attribute__((ext_vector_type(8)))  _Float16 v8h;
typedef __attribute__((ext_vector_type(8)))  float    v8f;
typedef int v4i_gnu __attribute__((vector_size(16)));

union AV { v16h v; v8h h[2]; };

#define S_LEN 2048
#define D_IN  1024
#define UHID  512
#define GATE  2048   // 4*U
#define BATCH 8

#if defined(__gfx1250__) && __has_builtin(__builtin_amdgcn_global_load_async_to_lds_b128)
#define HAS_ASYNC 1
#else
#define HAS_ASYNC 0
#endif

__device__ __forceinline__ float sigm(float x) { return 1.0f / (1.0f + __expf(-x)); }

__device__ __forceinline__ void wait_async0() {
#if HAS_ASYNC
#if __has_builtin(__builtin_amdgcn_s_wait_asynccnt)
    __builtin_amdgcn_s_wait_asynccnt(0);
#else
    asm volatile("s_wait_asynccnt 0x0" ::: "memory");
#endif
#endif
}

__device__ __forceinline__ void async_cp16(const _Float16* g, _Float16* l) {
#if HAS_ASYNC
    __builtin_amdgcn_global_load_async_to_lds_b128(
        (__attribute__((address_space(1))) v4i_gnu*)g,
        (__attribute__((address_space(3))) v4i_gnu*)l, 0, 0);
#endif
}

// ---------------------------------------------------------------------------
// f32 -> f16 elementwise convert (layer-0 input)
// ---------------------------------------------------------------------------
__global__ void cvt_f32_f16(const float* __restrict__ src, _Float16* __restrict__ dst, size_t n) {
    size_t i = (size_t)blockIdx.x * blockDim.x + threadIdx.x;
    if (i < n) dst[i] = (_Float16)src[i];
}

// ---------------------------------------------------------------------------
// Pack f32 [K][2048] weight matrices into the WMMA f16 B-operand lane layout:
// dst[((kt*128 + nt)*32 + lane)*16 + j], half=lane>>4: j<8: K=kt*32+8*half+j, j>=8: K=kt*32+16+8*half+(j-8),
// column n = nt*16 + (lane&15). One thread emits one lane's 16 contiguous f16.
// ---------------------------------------------------------------------------
__global__ void pack_w(const float* __restrict__ src, _Float16* __restrict__ dst,
                       int Ktiles, int nMats) {
    const int NT = 128;
    int gid = blockIdx.x * blockDim.x + threadIdx.x;
    int perMat = Ktiles * NT * 32;
    if (gid >= perMat * nMats) return;
    int mat = gid / perMat;
    int r   = gid % perMat;
    int kt  = r / (NT * 32);
    int r2  = r % (NT * 32);
    int nt  = r2 / 32;
    int lane = r2 % 32;
    int half = lane >> 4, ln = lane & 15;
    const float* s = src + (size_t)mat * (size_t)(Ktiles * 32) * 2048;
    _Float16*    d = dst + (size_t)mat * (size_t)perMat * 16 + (size_t)r * 16;
    int n = nt * 16 + ln;
    #pragma unroll
    for (int j = 0; j < 16; ++j) {
        int k = kt * 32 + (j < 8 ? half * 8 + j : 16 + half * 8 + (j - 8));
        d[j] = (_Float16)s[(size_t)k * 2048 + n];
    }
}

// ---------------------------------------------------------------------------
// Phase 1: xw[s][g][b] = sum_d y[b][s_src][d] * Wx[d][g] + bias[g]   (f16 out)
// One 16x16 output tile per wave: M covers (s0, b0..7),(s0+1, b0..7).
// ---------------------------------------------------------------------------
__global__ void gemm_xw(const _Float16* __restrict__ y,     // [8][2048][1024] f16
                        const _Float16* __restrict__ Wp,    // packed [32][128][32][16]
                        const float*    __restrict__ bias,  // [2048]
                        _Float16*       __restrict__ xw,    // [2048][2048][8] f16
                        int dir) {
    int wid  = (blockIdx.x * blockDim.x + threadIdx.x) >> 5;   // 0..131071
    int lane = threadIdx.x & 31;
    int st = wid >> 7;           // sequence pair 0..1023
    int nt = wid & 127;          // N tile 0..127
    int half = lane >> 4, ln = lane & 15;
    int s0 = st * 2;

    int rowM  = ln;                       // 0..15
    int s_int = s0 + (rowM >> 3);
    int b     = rowM & 7;
    int s_src = dir ? (S_LEN - 1 - s_int) : s_int;
    const _Float16* arow = y + ((size_t)b * S_LEN + s_src) * D_IN;

    int col = nt * 16 + ln;
    float bv = bias[col];
    v8f acc = { bv, bv, bv, bv, bv, bv, bv, bv };

    const _Float16* wp = Wp + ((size_t)nt * 32 + lane) * 16;
    const size_t ktStride = (size_t)128 * 32 * 16;

    #pragma unroll 4
    for (int kk = 0; kk < 32; ++kk) {
        int kb = kk * 32;
        AV a;
        a.h[0] = *(const v8h*)(arow + kb + half * 8);
        a.h[1] = *(const v8h*)(arow + kb + 16 + half * 8);
        v16h bm = *(const v16h*)(wp + (size_t)kk * ktStride);
        __builtin_prefetch(wp + (size_t)(kk + 1) * ktStride, 0, 1);
        acc = __builtin_amdgcn_wmma_f32_16x16x32_f16(false, a.v, false, bm,
                                                     (short)0, acc, false, false);
    }

    int s_out = s0 + half;
    _Float16* o = xw + ((size_t)s_out * GATE + col) * 8;
    v8h ov;
    #pragma unroll
    for (int j = 0; j < 8; ++j) ov[j] = (_Float16)acc[j];
    *(v8h*)o = ov;
}

// ---------------------------------------------------------------------------
// Phase 2: sequential LSTM recurrence, one workgroup per direction.
// 32 waves (1024 thr); wave w owns hidden units [16w,16w+16) => 4 gate tiles.
// h kept in LDS as zero-padded 16x512 f16 A-matrix; c in registers.
// xw gate pre-activations double-buffered in LDS via async-to-LDS prefetch
// (GLOBAL_LOAD_ASYNC_TO_LDS_B128 + s_wait_asynccnt), hiding HBM latency
// behind the WMMA K-loop of the previous timestep.
// ---------------------------------------------------------------------------
__global__ void __launch_bounds__(1024)
lstm_rec(const _Float16* __restrict__ xw,    // [2][2048][2048][8] f16
         const _Float16* __restrict__ Whp,   // packed, per dir 512*2048 f16
         _Float16*       __restrict__ ynext) // [8][2048][1024] f16
{
    int dir = blockIdx.x;
    const _Float16* xwd = xw  + (size_t)dir * (size_t)S_LEN * GATE * 8;
    const _Float16* wp0 = Whp + (size_t)dir * (size_t)UHID * GATE;

    extern __shared__ _Float16 smem[];
    _Float16* hbuf = smem;                        // 16*512 = 8192 f16 (16 KB)
    _Float16* xwb0 = smem + 16 * UHID;            // 2048*8 f16 (32 KB)
    _Float16* xwb1 = xwb0 + GATE * 8;             // 2048*8 f16 (32 KB)

    int tid  = threadIdx.x;
    int wave = tid >> 5, lane = tid & 31, half = lane >> 4, ln = lane & 15;
    int ubase = wave * 16;
    int u0 = ubase + ln;

    for (int i = tid; i < 16 * UHID; i += blockDim.x) hbuf[i] = (_Float16)0.0f;

#if HAS_ASYNC
    // preload xw slice for t = 0
    {
        const _Float16* g0 = xwd;
        #pragma unroll
        for (int k2 = 0; k2 < 2; ++k2) {
            int chunk = tid + k2 * 1024;          // 0..2047, 16B each
            async_cp16(g0 + (size_t)chunk * 8, xwb0 + (size_t)chunk * 8);
        }
    }
#endif

    const v8f vzero = {0, 0, 0, 0, 0, 0, 0, 0};
    v8f c = vzero;

    for (int t = 0; t < S_LEN; ++t) {
        wait_async0();
        __syncthreads();   // xw slice for t ready; h(t-1) visible

        const _Float16* xwt = (t & 1) ? xwb1 : xwb0;
#if HAS_ASYNC
        if (t + 1 < S_LEN) {   // prefetch next step while this one computes
            const _Float16* gn = xwd + (size_t)(t + 1) * GATE * 8;
            _Float16* dn = (t & 1) ? xwb0 : xwb1;
            #pragma unroll
            for (int k2 = 0; k2 < 2; ++k2) {
                int chunk = tid + k2 * 1024;
                async_cp16(gn + (size_t)chunk * 8, dn + (size_t)chunk * 8);
            }
        }
#endif

        v8f acc[4];
        // C init: z := xw[t]  (real batch rows only; pad rows stay 0)
        #pragma unroll
        for (int g = 0; g < 4; ++g) {
            if (half == 0) {
                int col = g * UHID + ubase + ln;
#if HAS_ASYNC
                v8h xv = *(const v8h*)(xwt + (size_t)col * 8);
#else
                v8h xv = *(const v8h*)(xwd + ((size_t)t * GATE + col) * 8);
#endif
                #pragma unroll
                for (int j = 0; j < 8; ++j) acc[g][j] = (float)xv[j];
            } else {
                acc[g] = vzero;
            }
        }

        // z += h @ Wh : K = 512, A fragment reused across the wave's 4 tiles
        #pragma unroll 1
        for (int kk = 0; kk < 16; ++kk) {
            int kb = kk * 32;
            AV a;
            a.h[0] = *(const v8h*)(hbuf + ln * UHID + kb + half * 8);
            a.h[1] = *(const v8h*)(hbuf + ln * UHID + kb + 16 + half * 8);
            #pragma unroll
            for (int g = 0; g < 4; ++g) {
                int nt = g * 32 + wave;
                v16h bm = *(const v16h*)(wp0 + ((size_t)(kk * 128 + nt) * 32 + lane) * 16);
                acc[g] = __builtin_amdgcn_wmma_f32_16x16x32_f16(
                    false, a.v, false, bm, (short)0, acc[g], false, false);
            }
        }

        // gates: i=acc[0], f=acc[1], g=acc[2], o=acc[3]
        v8f h;
        #pragma unroll
        for (int j = 0; j < 8; ++j) {
            float cn = sigm(acc[1][j]) * c[j] + sigm(acc[0][j]) * tanhf(acc[2][j]);
            c[j] = cn;
            h[j] = sigm(acc[3][j]) * tanhf(cn);
        }

        __syncthreads();   // all waves finished reading h(t-1) and xw slice
        #pragma unroll
        for (int j = 0; j < 8; ++j) {
            int m = j + half * 8;
            hbuf[m * UHID + u0] = (_Float16)h[j];
        }
        if (half == 0) {
            int pos = dir ? (S_LEN - 1 - t) : t;
            #pragma unroll
            for (int j = 0; j < 8; ++j) {
                size_t base = ((size_t)j * S_LEN + pos) * (2 * UHID) + (size_t)dir * UHID;
                ynext[base + u0] = (_Float16)h[j];
            }
        }
        // top-of-loop barrier of t+1 orders these writes against next reads
    }
}

// ---------------------------------------------------------------------------
// Phase 3: out[b,s] = y[b,s,:] . Wd + bd   (OUT == 1, memory-bound)
// ---------------------------------------------------------------------------
__global__ void dense_out(const _Float16* __restrict__ y, const float* __restrict__ Wd,
                          const float* __restrict__ bd, float* __restrict__ out) {
    int idx = blockIdx.x * blockDim.x + threadIdx.x;
    if (idx >= BATCH * S_LEN) return;
    const _Float16* row = y + (size_t)idx * (2 * UHID);
    float s = 0.0f;
    #pragma unroll 8
    for (int k = 0; k < 2 * UHID; ++k) s += (float)row[k] * Wd[k];
    out[idx] = s + bd[0];
}

// ---------------------------------------------------------------------------
extern "C" void kernel_launch(void* const* d_in, const int* in_sizes, int n_in,
                              void* d_out, int out_size, void* d_ws, size_t ws_size,
                              hipStream_t stream) {
    const float* x    = (const float*)d_in[0];  // [8,2048,1024]
    const float* Wx   = (const float*)d_in[1];  // [3,2,1024,2048]
    const float* Wh   = (const float*)d_in[2];  // [3,2,512,2048]
    const float* bias = (const float*)d_in[3];  // [3,2,2048]
    const float* Wd   = (const float*)d_in[4];  // [1024,1]
    const float* bd   = (const float*)d_in[5];  // [1]

    char* ws = (char*)d_ws;
    _Float16* yA  = (_Float16*)(ws);                       // 33,554,432 B
    _Float16* yB  = (_Float16*)(ws + 33554432);            // 33,554,432 B
    _Float16* WxP = (_Float16*)(ws + 67108864);            // 25,165,824 B
    _Float16* WhP = (_Float16*)(ws + 92274688);            // 12,582,912 B
    _Float16* XW  = (_Float16*)(ws + 104857600);           // 134,217,728 B

    const size_t wxMat = (size_t)1024 * 2048;  // packed elems per Wx matrix
    const size_t whMat = (size_t)512 * 2048;   // packed elems per Wh matrix
    const size_t ldsBytes = (size_t)(16 * UHID + 2 * GATE * 8) * sizeof(_Float16); // 80 KB

    cvt_f32_f16<<<65536, 256, 0, stream>>>(x, yA, (size_t)8 * 2048 * 1024);
    pack_w<<<3072, 256, 0, stream>>>(Wx, WxP, 32, 6);
    pack_w<<<1536, 256, 0, stream>>>(Wh, WhP, 16, 6);

    for (int l = 0; l < 3; ++l) {
        _Float16* yc  = (l & 1) ? yB : yA;
        _Float16* ynx = (l & 1) ? yA : yB;
        for (int dir = 0; dir < 2; ++dir) {
            gemm_xw<<<16384, 256, 0, stream>>>(
                yc, WxP + (size_t)(l * 2 + dir) * wxMat,
                bias + (size_t)(l * 2 + dir) * 2048,
                XW + (size_t)dir * (size_t)S_LEN * GATE * 8, dir);
        }
        lstm_rec<<<2, 1024, ldsBytes, stream>>>(XW, WhP + (size_t)l * 2 * whMat, ynx);
    }
    // after l=2, final activations are in yB
    dense_out<<<64, 256, 0, stream>>>(yB, Wd, bd, (float*)d_out);
}
